// CrossAttention_84112639525254
// MI455X (gfx1250) — matrix operand order repaired and checked by reading, not compiled, not run on hardware
//
#include <hip/hip_runtime.h>

typedef __attribute__((ext_vector_type(16))) __bf16 v16bf;
typedef __attribute__((ext_vector_type(8)))  float  v8f;
typedef __attribute__((ext_vector_type(4))) unsigned int u32x4;
typedef __attribute__((ext_vector_type(8))) int i32x8;
typedef __attribute__((ext_vector_type(4))) int i32x4;

#define SLEN  4096
#define CDIM  256
#define NHEADS 4
#define HDIM  64

union FragBF {
    v16bf v;
    __bf16 e[16];
    unsigned int u[8];
};

// ---- WMMA VGPR layout helpers (CDNA5 ISA §7.12.2, wave32) ----
// 16-bit A 16x32 (MxK): lane holds row M=lane&15; halves cover contiguous K pairs.
__device__ __forceinline__ int a_kbase(int lane, int v) {
    return ((v & 4) << 2) + ((lane >> 4) << 3) + ((v & 3) << 1);
}
// 16-bit B 32x16 (KxN): lane holds col N=lane&15; K = 16*(v>=4) + 8*(lane>=16) + 2*(v&3) + p
__device__ __forceinline__ int b_kbase(int lane, int v) {
    return ((v & 4) << 2) + ((lane >> 4) << 3) + ((v & 3) << 1);
}
// f32 C/D 16x16: row M = r + 8*(lane>=16), col N = lane&15

// ---- Tensor Data Mover: 2D tile Global -> LDS (CDNA5 ISA) ----
// data_size_code: 0=1B 1=2B 2=4B 3=8B.  tile rows are contiguous in LDS.
__device__ __forceinline__ void tdm_load_2d(
    unsigned lds_addr, const void* gptr,
    unsigned tile_d0, unsigned tile_d1,
    unsigned tensor_d0, unsigned tensor_d1,
    unsigned long long stride0, unsigned data_size_code)
{
    unsigned long long ga = (unsigned long long)(uintptr_t)gptr;
    u32x4 g0;
    g0.x = 1u;                                              // count=1 (valid D#)
    g0.y = lds_addr;                                        // LDS byte address
    g0.z = (unsigned)(ga & 0xffffffffull);                  // global_addr[31:0]
    g0.w = (unsigned)((ga >> 32) & 0x1ffffffull) | (2u << 30); // addr[56:32] | type=2
    i32x8 g1;
    g1[0] = (int)(data_size_code << 16);                    // wg_mask=0, data_size
    g1[1] = (int)((tensor_d0 & 0xffffu) << 16);             // tensor_dim0[15:0]
    g1[2] = (int)((tensor_d0 >> 16) | ((tensor_d1 & 0xffffu) << 16));
    g1[3] = (int)((tensor_d1 >> 16) | (tile_d0 << 16));     // tile_dim0
    g1[4] = (int)tile_d1;                                   // tile_dim1 (tile_dim2=0)
    g1[5] = (int)(unsigned)(stride0 & 0xffffffffull);       // tensor_dim0_stride
    g1[6] = (int)(unsigned)((stride0 >> 32) & 0xffffull);
    g1[7] = 0;
    i32x4 z4 = {};
#if defined(__clang_major__) && __clang_major__ >= 23
    i32x8 z8 = {};
    __builtin_amdgcn_tensor_load_to_lds(g0, g1, z4, z4, z8, 0);
#else
    __builtin_amdgcn_tensor_load_to_lds(g0, g1, z4, z4, 0);
#endif
}

// =====================================================================
// Kernel 1: QKV projection GEMM (M=256 outs, K=256 chans, N=16 spatial/block)
// Q,K stored bf16 as [b*4+head][s][64]; V stored bf16 as [b*4+head][64][s]
// =====================================================================
__global__ __launch_bounds__(256) void proj_qkv(
    const float* __restrict__ xq, const float* __restrict__ xkv,
    const float* __restrict__ wq, const float* __restrict__ bq,
    const float* __restrict__ wk, const float* __restrict__ bk,
    const float* __restrict__ wv, const float* __restrict__ bv,
    __bf16* __restrict__ Qg, __bf16* __restrict__ Kg, __bf16* __restrict__ Vg)
{
    const int which = blockIdx.z;            // 0=q 1=k 2=v
    const int b     = blockIdx.y;
    const int s0    = blockIdx.x * 16;
    const float* x    = (which == 0) ? xq : xkv;
    const float* wgt  = (which == 0) ? wq : (which == 1 ? wk : wv);
    const float* bias = (which == 0) ? bq : (which == 1 ? bk : bv);

    __shared__ __align__(16) __bf16 Bt[16][CDIM];   // [s][c] bf16 activation tile

    const int t = threadIdx.x;               // 0..255
    {   // cooperative load+convert: thread t owns channel c=t, 16 spatial values
        const float4* src = (const float4*)(x + ((size_t)b * CDIM + t) * SLEN + s0);
        #pragma unroll
        for (int i = 0; i < 4; ++i) {
            float4 f = src[i];
            Bt[i * 4 + 0][t] = (__bf16)f.x;
            Bt[i * 4 + 1][t] = (__bf16)f.y;
            Bt[i * 4 + 2][t] = (__bf16)f.z;
            Bt[i * 4 + 3][t] = (__bf16)f.w;
        }
    }
    __syncthreads();

    const int lane = t & 31, w = t >> 5;

    #pragma unroll
    for (int half = 0; half < 2; ++half) {
        const int ot = w + half * 8;                 // output 16-row tile id (0..15)
        v8f acc = {};
        const int arow = ot * 16 + (lane & 15);
        const float2* wrow = (const float2*)(wgt + (size_t)arow * CDIM);
        #pragma unroll
        for (int kc = 0; kc < 8; ++kc) {
            FragBF a, bb;
            #pragma unroll
            for (int v = 0; v < 8; ++v) {
                float2 f = wrow[(kc * 32 + a_kbase(lane, v)) >> 1];
                a.e[2 * v + 0] = (__bf16)f.x;
                a.e[2 * v + 1] = (__bf16)f.y;
            }
            const int sn = lane & 15;
            #pragma unroll
            for (int v = 0; v < 8; ++v)
                bb.u[v] = *(const unsigned int*)&Bt[sn][kc * 32 + b_kbase(lane, v)];
            acc = __builtin_amdgcn_wmma_f32_16x16x32_bf16(
                false, a.v, false, bb.v, (short)0, acc, false, false);
        }
        #pragma unroll
        for (int r = 0; r < 8; ++r) {
            const int o = ot * 16 + r + ((lane >> 4) << 3);
            const int s = s0 + (lane & 15);
            const __bf16 val = (__bf16)(acc[r] + bias[o]);
            const int head = o >> 6, d = o & 63;
            const size_t bh = (size_t)b * NHEADS + head;
            if (which == 0)      Qg[(bh * SLEN + s) * HDIM + d] = val;
            else if (which == 1) Kg[(bh * SLEN + s) * HDIM + d] = val;
            else                 Vg[(bh * HDIM + d) * SLEN + s] = val;
        }
    }
}

// =====================================================================
// Kernel 2: flash attention with TDM double-buffered K/V tile staging.
// Block=256 (8 waves); wave owns a 16-row query tile; 64-wide key tiles.
// Output bf16 [b][s][c] (coalesced epilogue).
// =====================================================================
__global__ __launch_bounds__(256) void attn_fwd(
    const __bf16* __restrict__ Qg, const __bf16* __restrict__ Kg,
    const __bf16* __restrict__ Vg, __bf16* __restrict__ Og)
{
    const int bh   = blockIdx.y;             // 0..7
    const int b    = bh >> 2, head = bh & 3;
    const int t    = threadIdx.x, lane = t & 31, w = t >> 5;
    const int i0   = (blockIdx.x * 8 + w) * 16;
    const float scale = 0.125f;              // HEAD_DIM^-0.5

    __shared__ __align__(16) __bf16 Kt[2][64 * 64];     // [j][d] ping-pong 16KB
    __shared__ __align__(16) __bf16 Vt[2][64 * 64];     // [d][j] ping-pong 16KB
    __shared__ __align__(16) __bf16 Pt[8][16 * 64];     // per-wave P tile 16KB

    // Q fragments (16 rows x 64 d) held in registers for the whole loop
    FragBF qa[2];
    {
        const __bf16* qrow = Qg + ((size_t)bh * SLEN + i0 + (lane & 15)) * HDIM;
        #pragma unroll
        for (int kc = 0; kc < 2; ++kc)
            #pragma unroll
            for (int v = 0; v < 8; ++v)
                qa[kc].u[v] = *(const unsigned int*)(qrow + kc * 32 + a_kbase(lane, v));
    }

    v8f out[4] = {};
    float m[8], lsum[8];
    #pragma unroll
    for (int r = 0; r < 8; ++r) { m[r] = -INFINITY; lsum[r] = 0.0f; }

    const unsigned ldsK[2] = { (unsigned)(uintptr_t)&Kt[0][0], (unsigned)(uintptr_t)&Kt[1][0] };
    const unsigned ldsV[2] = { (unsigned)(uintptr_t)&Vt[0][0], (unsigned)(uintptr_t)&Vt[1][0] };

    const int NT = SLEN / 64;
    if (w == 0) {   // preload tile 0 (TDM, per-wave op, EXEC ignored)
        tdm_load_2d(ldsK[0], Kg + ((size_t)bh * SLEN) * HDIM, 64, 64, 64, SLEN,
                    64ull, 1u);
        tdm_load_2d(ldsV[0], Vg + (size_t)bh * HDIM * SLEN,   64, 64, SLEN, HDIM,
                    (unsigned long long)SLEN, 1u);
    }

    for (int it = 0; it < NT; ++it) {
        const int buf = it & 1;
        __syncthreads();                     // readers of buf^1 finished
        if (w == 0) {
            if (it + 1 < NT) {
                const int j1 = (it + 1) * 64;
                tdm_load_2d(ldsK[buf ^ 1], Kg + ((size_t)bh * SLEN + j1) * HDIM,
                            64, 64, 64, SLEN, 64ull, 1u);
                tdm_load_2d(ldsV[buf ^ 1], Vg + (size_t)bh * HDIM * SLEN + j1,
                            64, 64, SLEN, HDIM, (unsigned long long)SLEN, 1u);
                __builtin_amdgcn_s_wait_tensorcnt((short)2);  // oldest pair done
            } else {
                __builtin_amdgcn_s_wait_tensorcnt((short)0);
            }
        }
        __syncthreads();                     // tile `it` visible to all waves
        const __bf16* Ktb = Kt[buf];
        const __bf16* Vtb = Vt[buf];

        // ---- S = Q K^T  (16 x 64 scores) ----
        v8f sc[4];
        #pragma unroll
        for (int jn = 0; jn < 4; ++jn) {
            sc[jn] = (v8f){};
            const int jl = jn * 16 + (lane & 15);
            #pragma unroll
            for (int kc = 0; kc < 2; ++kc) {
                FragBF kb;
                #pragma unroll
                for (int v = 0; v < 8; ++v)
                    kb.u[v] = *(const unsigned int*)(Ktb + jl * 64 + kc * 32 + b_kbase(lane, v));
                sc[jn] = __builtin_amdgcn_wmma_f32_16x16x32_bf16(
                    false, qa[kc].v, false, kb.v, (short)0, sc[jn], false, false);
            }
        }

        // ---- online softmax (row reductions across 16 lanes of each half) ----
        #pragma unroll
        for (int r = 0; r < 8; ++r) {
            float mx = fmaxf(fmaxf(sc[0][r], sc[1][r]), fmaxf(sc[2][r], sc[3][r]));
            #pragma unroll
            for (int off = 1; off < 16; off <<= 1)
                mx = fmaxf(mx, __shfl_xor(mx, off, 32));
            const float mnew = fmaxf(m[r], mx * scale);
            const float corr = __expf(m[r] - mnew);
            float rs = 0.0f;
            #pragma unroll
            for (int jn = 0; jn < 4; ++jn) {
                const float p = __expf(sc[jn][r] * scale - mnew);
                sc[jn][r] = p;
                rs += p;
            }
            #pragma unroll
            for (int off = 1; off < 16; off <<= 1)
                rs += __shfl_xor(rs, off, 32);
            lsum[r] = lsum[r] * corr + rs;
            m[r] = mnew;
            #pragma unroll
            for (int db = 0; db < 4; ++db) out[db][r] *= corr;
        }

        // ---- re-layout P: C/D layout -> A layout via per-wave LDS ----
        #pragma unroll
        for (int r = 0; r < 8; ++r) {
            const int il = r + ((lane >> 4) << 3);
            #pragma unroll
            for (int jn = 0; jn < 4; ++jn)
                Pt[w][il * 64 + jn * 16 + (lane & 15)] = (__bf16)sc[jn][r];
        }
        asm volatile("s_wait_dscnt 0" ::: "memory");   // in-wave store->load

        // ---- O += P V ----
        #pragma unroll
        for (int kc = 0; kc < 2; ++kc) {
            FragBF pa;
            const int il = lane & 15;
            #pragma unroll
            for (int v = 0; v < 8; ++v)
                pa.u[v] = *(const unsigned int*)(&Pt[w][il * 64 + kc * 32 + a_kbase(lane, v)]);
            #pragma unroll
            for (int db = 0; db < 4; ++db) {
                FragBF vb;
                const int dn = db * 16 + (lane & 15);
                #pragma unroll
                for (int v = 0; v < 8; ++v)
                    vb.u[v] = *(const unsigned int*)(Vtb + dn * 64 + kc * 32 + b_kbase(lane, v));
                out[db] = __builtin_amdgcn_wmma_f32_16x16x32_bf16(
                    false, pa.v, false, vb.v, (short)0, out[db], false, false);
            }
        }
    }

    // ---- epilogue: normalize, write bf16 [b][s][c] (coalesced in c) ----
    #pragma unroll
    for (int r = 0; r < 8; ++r) {
        const float inv = 1.0f / lsum[r];
        const int i = i0 + r + ((lane >> 4) << 3);
        #pragma unroll
        for (int db = 0; db < 4; ++db) {
            const int c = head * HDIM + db * 16 + (lane & 15);
            Og[((size_t)b * SLEN + i) * CDIM + c] = (__bf16)(out[db][r] * inv);
        }
    }
}

// =====================================================================
// Kernel 3: output projection GEMM; activation tile staged by TDM (contiguous
// 16x256 bf16 tile); fp32 result + bias -> d_out [b][c][h][w]
// =====================================================================
__global__ __launch_bounds__(256) void proj_out(
    const __bf16* __restrict__ Og, const float* __restrict__ wo,
    const float* __restrict__ bo, float* __restrict__ out)
{
    const int b  = blockIdx.y;
    const int s0 = blockIdx.x * 16;

    __shared__ __align__(16) __bf16 Bt[16][CDIM];      // [s][c], 8KB contiguous

    const int t = threadIdx.x, lane = t & 31, w = t >> 5;
    if (w == 0) {   // one TDM copy: whole tile is contiguous in [b][s][c] layout
        tdm_load_2d((unsigned)(uintptr_t)&Bt[0][0],
                    Og + ((size_t)b * SLEN + s0) * CDIM,
                    4096u, 1u, 4096u, 1u, 4096ull, 1u);
        __builtin_amdgcn_s_wait_tensorcnt((short)0);
    }
    __syncthreads();

    #pragma unroll
    for (int half = 0; half < 2; ++half) {
        const int ot = w + half * 8;
        v8f acc = {};
        const int arow = ot * 16 + (lane & 15);
        const float2* wrow = (const float2*)(wo + (size_t)arow * CDIM);
        #pragma unroll
        for (int kc = 0; kc < 8; ++kc) {
            FragBF a, bb;
            #pragma unroll
            for (int v = 0; v < 8; ++v) {
                float2 f = wrow[(kc * 32 + a_kbase(lane, v)) >> 1];
                a.e[2 * v + 0] = (__bf16)f.x;
                a.e[2 * v + 1] = (__bf16)f.y;
            }
            const int sn = lane & 15;
            #pragma unroll
            for (int v = 0; v < 8; ++v)
                bb.u[v] = *(const unsigned int*)&Bt[sn][kc * 32 + b_kbase(lane, v)];
            acc = __builtin_amdgcn_wmma_f32_16x16x32_bf16(
                false, a.v, false, bb.v, (short)0, acc, false, false);
        }
        #pragma unroll
        for (int r = 0; r < 8; ++r) {
            const int o = ot * 16 + r + ((lane >> 4) << 3);
            const int s = s0 + (lane & 15);
            out[((size_t)b * CDIM + o) * SLEN + s] = acc[r] + bo[o];
        }
    }
}

// =====================================================================
extern "C" void kernel_launch(void* const* d_in, const int* in_sizes, int n_in,
                              void* d_out, int out_size, void* d_ws, size_t ws_size,
                              hipStream_t stream) {
    (void)in_sizes; (void)n_in; (void)out_size; (void)ws_size;
    const float* xq  = (const float*)d_in[0];
    const float* xkv = (const float*)d_in[1];
    const float* wq  = (const float*)d_in[2];
    const float* bq  = (const float*)d_in[3];
    const float* wk  = (const float*)d_in[4];
    const float* bk  = (const float*)d_in[5];
    const float* wv  = (const float*)d_in[6];
    const float* bv  = (const float*)d_in[7];
    const float* wo  = (const float*)d_in[8];
    const float* bo  = (const float*)d_in[9];
    float* out = (float*)d_out;

    char* ws = (char*)d_ws;
    const size_t qkv_bytes = (size_t)2 * NHEADS * SLEN * HDIM * sizeof(__bf16); // 4 MB
    __bf16* Qg = (__bf16*)(ws);
    __bf16* Kg = (__bf16*)(ws + qkv_bytes);
    __bf16* Vg = (__bf16*)(ws + 2 * qkv_bytes);
    __bf16* Og = (__bf16*)(ws + 3 * qkv_bytes);          // 4 MB  (total 16 MB)

    proj_qkv<<<dim3(SLEN / 16, 2, 3), 256, 0, stream>>>(
        xq, xkv, wq, bq, wk, bk, wv, bv, Qg, Kg, Vg);
    attn_fwd<<<dim3(SLEN / 128, 8), 256, 0, stream>>>(Qg, Kg, Vg, Og);
    proj_out<<<dim3(SLEN / 16, 2), 256, 0, stream>>>(Og, wo, bo, out);
}
